// GPT_11931419149000
// MI455X (gfx1250) — compile-verified, matrix operand order checked by reading, not run on hardware
//
#include <hip/hip_runtime.h>
#include <hip/hip_bf16.h>
#include <math.h>

// ---------------------------------------------------------------------------
// Model constants (from the reference)
// ---------------------------------------------------------------------------
#define LL 4
#define HH 16
#define DD 1024
#define TT 1024
#define BB 8
#define MM 134
#define BT (BB * TT)          // 8192 token rows
#define HD 64                 // head dim

typedef __bf16 bf16;
typedef __attribute__((ext_vector_type(8)))  bf16  bf16x8;
typedef __attribute__((ext_vector_type(16))) bf16  bf16x16;
typedef __attribute__((ext_vector_type(8)))  float f32x8;
typedef __attribute__((ext_vector_type(4)))  unsigned int u32x4;
typedef __attribute__((ext_vector_type(8)))  int i32x8;
typedef __attribute__((ext_vector_type(4)))  int i32x4;

// round-to-nearest-even f32 -> bf16
__device__ __forceinline__ bf16 f2bf(float f) {
    unsigned u = __builtin_bit_cast(unsigned, f);
    unsigned r = (u + 0x7FFFu + ((u >> 16) & 1u)) >> 16;
    return __builtin_bit_cast(bf16, (unsigned short)r);
}

__device__ __forceinline__ bf16x16 cat8(bf16x8 lo, bf16x8 hi) {
    return __builtin_shufflevector(lo, hi, 0,1,2,3,4,5,6,7,8,9,10,11,12,13,14,15);
}

__device__ __forceinline__ f32x8 wmma_bf16(bf16x16 a, bf16x16 b, f32x8 c) {
    return __builtin_amdgcn_wmma_f32_16x16x32_bf16(
        /*neg_a=*/false, a, /*neg_b=*/false, b,
        /*c_mod=*/(short)0, c, /*reuse_a=*/false, /*reuse_b=*/false);
}

// ---------------------------------------------------------------------------
// TDM: DMA a 2-D bf16 tile [tile1 rows x 32 cols] (row stride = strideElems)
// from global memory into LDS, with 16B padding after each 64B row so the
// LDS row stride is 40 halves (bank-conflict-free ds_load_b128 fragments).
// D# layout per CDNA5 ISA. Issue from ONE wave only.
// This toolchain exposes the 6-arg builtin (g0,g1,g2,g3,g4,cpol).
// ---------------------------------------------------------------------------
__device__ __forceinline__ void tdm_load_2d(unsigned lds_addr,
                                            const bf16* gptr,
                                            unsigned tile1,
                                            unsigned long long strideElems) {
    const unsigned long long ga = (unsigned long long)(uintptr_t)gptr;
    u32x4 g0;
    g0[0] = 1u;                                   // count=1, user mode
    g0[1] = lds_addr;                             // LDS byte offset
    g0[2] = (unsigned)ga;                         // global_addr[31:0]
    g0[3] = (unsigned)(ga >> 32) | 0x80000000u;   // global_addr[56:32] | type=2
    i32x8 g1;
    // mask=0 | data_size=2B (1<<16) | pad_enable (1<<20) |
    // pad_interval=3 -> every 16 DWORDs (64B) | pad_amount=3 -> 4 DWORDs (16B)
    g1[0] = (int)((1u << 16) | (1u << 20) | (3u << 22) | (3u << 25));
    const unsigned td0 = 1u << 20;                // tensor_dim0 (no OOB)
    const unsigned td1 = 1u << 20;                // tensor_dim1 (no OOB)
    g1[1] = (int)((td0 & 0xFFFFu) << 16);
    g1[2] = (int)(((td0 >> 16) & 0xFFFFu) | ((td1 & 0xFFFFu) << 16));
    g1[3] = (int)(((td1 >> 16) & 0xFFFFu) | (32u << 16));       // tile_dim0=32
    g1[4] = (int)(tile1 & 0xFFFFu);                             // tile_dim1
    g1[5] = (int)(unsigned)(strideElems & 0xFFFFFFFFull);       // dim0_stride lo
    g1[6] = (int)(unsigned)((strideElems >> 32) & 0xFFFFull);   // dim0_stride hi
    g1[7] = 0;
    const i32x4 z4 = {0, 0, 0, 0};                // groups 2/3: 2-D tensor
    const i32x8 z8 = {0, 0, 0, 0, 0, 0, 0, 0};    // trailing group (unused)
    __builtin_amdgcn_tensor_load_to_lds(g0, g1, z4, z4, z8, 0);
}

// ---------------------------------------------------------------------------
// x = X + wpe   (wpe broadcast over batch; T*D is a power of two)
// ---------------------------------------------------------------------------
__global__ __launch_bounds__(256) void k_add_wpe(const float* __restrict__ X,
                                                 const float* __restrict__ wpe,
                                                 float* __restrict__ x) {
    size_t i = (size_t)blockIdx.x * 256 + threadIdx.x;
    x[i] = X[i] + wpe[i & ((size_t)TT * DD - 1)];
}

// ---------------------------------------------------------------------------
// Transposing weight convert: W[K,N] f32 -> Wt[N,K] bf16
// (reads coalesced; done once per launch, bandwidth-trivial)
// ---------------------------------------------------------------------------
__global__ __launch_bounds__(256) void k_cvt_t(const float* __restrict__ W,
                                               bf16* __restrict__ Wt,
                                               int Nn, int Kk) {
    size_t i = (size_t)blockIdx.x * 256 + threadIdx.x;
    if (i >= (size_t)Nn * Kk) return;
    int k = (int)(i / Nn), n = (int)(i % Nn);
    Wt[(size_t)n * Kk + k] = f2bf(W[i]);
}

// ---------------------------------------------------------------------------
// LayerNorm over D=1024, one block (256 thr) per row.
// PRERELU: x = relu(x) in-place first (stats over relu'd values).
// OUTF32 : write f32 output, else bf16.
// ---------------------------------------------------------------------------
template<bool PRERELU, bool OUTF32>
__global__ __launch_bounds__(256) void k_ln(float* __restrict__ x,
                                            const float* __restrict__ w,
                                            const float* __restrict__ b,
                                            bf16* __restrict__ outB,
                                            float* __restrict__ outF) {
    __shared__ float s1[256], s2[256];
    const int tid = threadIdx.x;
    const size_t row = blockIdx.x;
    float4 v = *(const float4*)(x + row * DD + tid * 4);
    if (PRERELU) {
        v.x = fmaxf(v.x, 0.f); v.y = fmaxf(v.y, 0.f);
        v.z = fmaxf(v.z, 0.f); v.w = fmaxf(v.w, 0.f);
        *(float4*)(x + row * DD + tid * 4) = v;
    }
    s1[tid] = v.x + v.y + v.z + v.w;
    s2[tid] = v.x*v.x + v.y*v.y + v.z*v.z + v.w*v.w;
    __syncthreads();
    for (int st = 128; st > 0; st >>= 1) {
        if (tid < st) { s1[tid] += s1[tid + st]; s2[tid] += s2[tid + st]; }
        __syncthreads();
    }
    const float mu  = s1[0] * (1.f / DD);
    const float var = s2[0] * (1.f / DD) - mu * mu;
    const float rs  = rsqrtf(var + 1e-5f);
    const int d = tid * 4;
    float4 wv = *(const float4*)(w + d);
    float4 bv = *(const float4*)(b + d);
    float o0 = (v.x - mu) * rs * wv.x + bv.x;
    float o1 = (v.y - mu) * rs * wv.y + bv.y;
    float o2 = (v.z - mu) * rs * wv.z + bv.z;
    float o3 = (v.w - mu) * rs * wv.w + bv.w;
    if (OUTF32) {
        float4 o = {o0, o1, o2, o3};
        *(float4*)(outF + row * DD + d) = o;
    } else {
        bf16* p = outB + row * DD + d;
        p[0] = f2bf(o0); p[1] = f2bf(o1); p[2] = f2bf(o2); p[3] = f2bf(o3);
    }
}

// ---------------------------------------------------------------------------
// WMMA GEMM: C[M,N] = A[M,K] (bf16, row-major) * Wt[N,K]^T (bf16) + bias
//   block: 256 thr = 8 waves; block tile 128x128; wave tile 32x64 (2x4 wmma)
//   A and B tiles staged by the Tensor Data Mover (double buffered).
//   M % 128 == 0, N % 128 == 0, K % 32 == 0   (all call sites satisfy this)
// Epilogue: optional ReLU, optional f32 residual add, f32 and/or bf16 store.
// ---------------------------------------------------------------------------
template<bool RELU, bool RESID, bool OUTF32, bool OUTBF16>
__global__ __launch_bounds__(256) void k_gemm(const bf16* __restrict__ A,
                                              const bf16* __restrict__ Wt,
                                              const float* __restrict__ bias,
                                              const float* __restrict__ resid,
                                              float* __restrict__ outF,
                                              bf16* __restrict__ outB,
                                              int Nn, int Kk) {
    __shared__ bf16 Asl[2][128 * 40];   // 128 rows x (32 + 8 pad) halves
    __shared__ bf16 Bsl[2][128 * 40];
    const int tid  = threadIdx.x;
    const int lane = tid & 31, wv = tid >> 5;
    const int wm = wv & 3, wn = wv >> 2;           // 4x2 wave grid
    const int l15 = lane & 15, hi = lane >> 4;
    const int kb = hi * 8;
    const int n0 = blockIdx.x * 128;
    const int m0 = blockIdx.y * 128;
    const bf16* Abase = A  + (size_t)m0 * Kk;
    const bf16* Bbase = Wt + (size_t)n0 * Kk;

    f32x8 acc[2][4];
    #pragma unroll
    for (int mt = 0; mt < 2; ++mt)
        #pragma unroll
        for (int nt = 0; nt < 4; ++nt)
            acc[mt][nt] = (f32x8){0,0,0,0,0,0,0,0};

    const int nk = Kk >> 5;
    if (wv == 0) {   // wave 0 drives the TDM
        tdm_load_2d((unsigned)(uintptr_t)&Asl[0][0], Abase, 128, (unsigned long long)Kk);
        tdm_load_2d((unsigned)(uintptr_t)&Bsl[0][0], Bbase, 128, (unsigned long long)Kk);
    }
    for (int kt = 0; kt < nk; ++kt) {
        const int cur = kt & 1;
        if (wv == 0) {
            if (kt + 1 < nk) {   // prefetch next K-slab into the other buffer
                tdm_load_2d((unsigned)(uintptr_t)&Asl[cur ^ 1][0],
                            Abase + (size_t)(kt + 1) * 32, 128, (unsigned long long)Kk);
                tdm_load_2d((unsigned)(uintptr_t)&Bsl[cur ^ 1][0],
                            Bbase + (size_t)(kt + 1) * 32, 128, (unsigned long long)Kk);
                __builtin_amdgcn_s_wait_tensorcnt((short)2);  // current pair landed
            } else {
                __builtin_amdgcn_s_wait_tensorcnt((short)0);
            }
        }
        __syncthreads();

        bf16x16 a[2], b[4];
        #pragma unroll
        for (int mt = 0; mt < 2; ++mt) {
            const bf16* r = &Asl[cur][(wm * 32 + mt * 16 + l15) * 40];
            a[mt] = cat8(*(const bf16x8*)(r + kb), *(const bf16x8*)(r + 16 + kb));
        }
        #pragma unroll
        for (int nt = 0; nt < 4; ++nt) {
            const bf16* r = &Bsl[cur][(wn * 64 + nt * 16 + l15) * 40];
            b[nt] = cat8(*(const bf16x8*)(r + kb), *(const bf16x8*)(r + 16 + kb));
        }
        #pragma unroll
        for (int mt = 0; mt < 2; ++mt)
            #pragma unroll
            for (int nt = 0; nt < 4; ++nt)
                acc[mt][nt] = wmma_bf16(a[mt], b[nt], acc[mt][nt]);
        __syncthreads();
    }

    // epilogue
    #pragma unroll
    for (int mt = 0; mt < 2; ++mt)
        #pragma unroll
        for (int nt = 0; nt < 4; ++nt) {
            const int n = n0 + wn * 64 + nt * 16 + l15;
            const float bv = bias ? bias[n] : 0.f;
            #pragma unroll
            for (int r = 0; r < 8; ++r) {
                const int m = m0 + wm * 32 + mt * 16 + hi * 8 + r;
                const size_t idx = (size_t)m * Nn + n;
                float f = acc[mt][nt][r] + bv;
                if (RELU)  f = fmaxf(f, 0.f);
                if (RESID) f += resid[idx];
                if (OUTF32)  outF[idx] = f;
                if (OUTBF16) outB[idx] = f2bf(f);
            }
        }
}

// ---------------------------------------------------------------------------
// Split qkv [B,T,3D] (bf16) into q,k [B,H,T,hd] and v-transposed [B,H,hd,T]
// ---------------------------------------------------------------------------
__global__ __launch_bounds__(256) void k_split_qkv(const bf16* __restrict__ qkv,
                                                   bf16* __restrict__ q,
                                                   bf16* __restrict__ k,
                                                   bf16* __restrict__ vt) {
    size_t i = (size_t)blockIdx.x * 256 + threadIdx.x;   // over B*T*D
    int d = (int)(i & (DD - 1));
    int t = (int)((i >> 10) & (TT - 1));
    int b = (int)(i >> 20);
    int h = d >> 6, hd = d & 63;
    const size_t base = ((size_t)b * TT + t) * (3 * DD);
    size_t qk_idx = (((size_t)(b * HH + h)) * TT + t) * HD + hd;
    q[qk_idx] = qkv[base + d];
    k[qk_idx] = qkv[base + DD + d];
    vt[(((size_t)(b * HH + h)) * HD + hd) * TT + t] = qkv[base + 2 * DD + d];
}

// ---------------------------------------------------------------------------
// Causal ReLU-attention, flash style. One wave per (b,h, 32-query block).
// blockDim = 128 (4 waves). y written as [B,T,D] bf16.
// ---------------------------------------------------------------------------
__global__ __launch_bounds__(128) void k_attn(const bf16* __restrict__ q,
                                              const bf16* __restrict__ k,
                                              const bf16* __restrict__ vt,
                                              bf16* __restrict__ y) {
    __shared__ bf16 sl[4][32 * 40];            // wave-private 32x32 S tile (+pad)
    const int lane = threadIdx.x & 31, wv = threadIdx.x >> 5;
    const int gw = blockIdx.x * 4 + wv;        // 0..4095
    const int qb = (gw & 31) * 32;             // query block start
    const int bh = gw >> 5;                    // 0..127  (b*H + h)
    const int l15 = lane & 15, hi = lane >> 4;
    const int kb = hi * 8;

    const bf16* qp = q  + (size_t)bh * TT * HD;
    const bf16* kp = k  + (size_t)bh * TT * HD;
    const bf16* vp = vt + (size_t)bh * HD * TT;

    // Q fragments: 2 M-tiles x 2 K-chunks (hd = 64)
    bf16x16 qa[2][2];
    #pragma unroll
    for (int mt = 0; mt < 2; ++mt) {
        const bf16* qr = qp + (size_t)(qb + mt * 16 + l15) * HD;
        #pragma unroll
        for (int ks = 0; ks < 2; ++ks) {
            bf16x8 lo = *(const bf16x8*)(qr + ks * 32 + kb);
            bf16x8 hx = *(const bf16x8*)(qr + ks * 32 + 16 + kb);
            qa[mt][ks] = cat8(lo, hx);
        }
    }

    f32x8 acc[2][4];
    #pragma unroll
    for (int mt = 0; mt < 2; ++mt)
        #pragma unroll
        for (int nt = 0; nt < 4; ++nt)
            acc[mt][nt] = (f32x8){0,0,0,0,0,0,0,0};

    bf16* sm = sl[wv];
    const int nkt = (qb >> 5) + 1;             // causal: key tiles <= query block
    for (int kt0 = 0; kt0 < nkt; ++kt0) {
        const int kt = kt0 * 32;
        // ---- S = Q * K^T  (2x2 tiles, K-dim 64) ----
        f32x8 s[2][2];
        #pragma unroll
        for (int mt = 0; mt < 2; ++mt)
            #pragma unroll
            for (int nt = 0; nt < 2; ++nt)
                s[mt][nt] = (f32x8){0,0,0,0,0,0,0,0};
        bf16x16 bk[2][2];
        #pragma unroll
        for (int nt = 0; nt < 2; ++nt) {
            const bf16* kr = kp + (size_t)(kt + nt * 16 + l15) * HD;
            #pragma unroll
            for (int ks = 0; ks < 2; ++ks) {
                bf16x8 lo = *(const bf16x8*)(kr + ks * 32 + kb);
                bf16x8 hx = *(const bf16x8*)(kr + ks * 32 + 16 + kb);
                bk[nt][ks] = cat8(lo, hx);
            }
        }
        #pragma unroll
        for (int mt = 0; mt < 2; ++mt)
            #pragma unroll
            for (int nt = 0; nt < 2; ++nt)
                #pragma unroll
                for (int ks = 0; ks < 2; ++ks)
                    s[mt][nt] = wmma_bf16(qa[mt][ks], bk[nt][ks], s[mt][nt]);

        // ---- causal mask + relu + scale; D-layout -> LDS (A-layout source) ----
        #pragma unroll
        for (int mt = 0; mt < 2; ++mt)
            #pragma unroll
            for (int nt = 0; nt < 2; ++nt) {
                const int mrow = mt * 16 + hi * 8;
                const int ncol = nt * 16 + l15;
                const int keyg = kt + ncol;
                #pragma unroll
                for (int r = 0; r < 8; ++r) {
                    const int qg = qb + mrow + r;
                    float v = (keyg <= qg) ? fmaxf(s[mt][nt][r], 0.f) * 0.125f : 0.f;
                    sm[(mrow + r) * 40 + ncol] = f2bf(v);
                }
            }
        // wave-private tile; DS ops are in-order within a wave — no barrier.

        // ---- Y += S * V   (K-dim = 32 keys) ----
        bf16x16 sa[2];
        #pragma unroll
        for (int mt = 0; mt < 2; ++mt) {
            const int row = mt * 16 + l15;
            bf16x8 lo = *(const bf16x8*)(sm + row * 40 + kb);
            bf16x8 hx = *(const bf16x8*)(sm + row * 40 + 16 + kb);
            sa[mt] = cat8(lo, hx);
        }
        bf16x16 bv[4];
        #pragma unroll
        for (int nt = 0; nt < 4; ++nt) {
            const bf16* vr = vp + (size_t)(nt * 16 + l15) * TT + kt;
            bf16x8 lo = *(const bf16x8*)(vr + kb);
            bf16x8 hx = *(const bf16x8*)(vr + 16 + kb);
            bv[nt] = cat8(lo, hx);
        }
        #pragma unroll
        for (int mt = 0; mt < 2; ++mt)
            #pragma unroll
            for (int nt = 0; nt < 4; ++nt)
                acc[mt][nt] = wmma_bf16(sa[mt], bv[nt], acc[mt][nt]);
    }

    // store y into [B,T,D] layout (merge heads)
    const int b = bh >> 4, h = bh & 15;
    #pragma unroll
    for (int mt = 0; mt < 2; ++mt)
        #pragma unroll
        for (int nt = 0; nt < 4; ++nt) {
            const int col = h * HD + nt * 16 + l15;
            #pragma unroll
            for (int r = 0; r < 8; ++r) {
                const int qg = qb + mt * 16 + hi * 8 + r;
                y[((size_t)b * TT + qg) * DD + col] = f2bf(acc[mt][nt][r]);
            }
        }
}

// ---------------------------------------------------------------------------
// emb[b,d] = mean_t relu(lnf[b,t,d])
// ---------------------------------------------------------------------------
__global__ __launch_bounds__(256) void k_final(const float* __restrict__ lnf,
                                               float* __restrict__ emb) {
    const int d = blockIdx.x * 256 + threadIdx.x;
    const int b = blockIdx.y;
    const float* p = lnf + (size_t)b * TT * DD + d;
    float s = 0.f;
    for (int t = 0; t < TT; ++t) s += fmaxf(p[(size_t)t * DD], 0.f);
    emb[(size_t)b * DD + d] = s * (1.f / TT);
}

// ---------------------------------------------------------------------------
// logits = relu(emb @ head_w + head_b); loss = 50*sqrt(mean((logits-Y)^2))
// ---------------------------------------------------------------------------
__global__ __launch_bounds__(256) void k_head(const float* __restrict__ emb,
                                              const float* __restrict__ hw,
                                              const float* __restrict__ hb,
                                              const float* __restrict__ Y,
                                              float* __restrict__ loss_out) {
    __shared__ float red[256];
    float sq = 0.f;
    for (int i = threadIdx.x; i < BB * MM; i += 256) {
        const int b = i / MM, m = i % MM;
        float a = hb[m];
        const float* e = emb + (size_t)b * DD;
        for (int d = 0; d < DD; ++d) a += e[d] * hw[(size_t)d * MM + m];
        const float df = fmaxf(a, 0.f) - Y[i];
        sq += df * df;
    }
    red[threadIdx.x] = sq;
    __syncthreads();
    for (int st = 128; st > 0; st >>= 1) {
        if (threadIdx.x < st) red[threadIdx.x] += red[threadIdx.x + st];
        __syncthreads();
    }
    if (threadIdx.x == 0)
        loss_out[0] = 50.f * sqrtf(red[0] / (float)(BB * MM));
}

// ---------------------------------------------------------------------------
// Host-side orchestration
// ---------------------------------------------------------------------------
extern "C" void kernel_launch(void* const* d_in, const int* in_sizes, int n_in,
                              void* d_out, int out_size, void* d_ws, size_t ws_size,
                              hipStream_t stream) {
    (void)in_sizes; (void)n_in; (void)out_size; (void)ws_size;
    const float* X       = (const float*)d_in[0];
    const float* Y       = (const float*)d_in[1];
    const float* wpe     = (const float*)d_in[2];
    const float* ln1_w   = (const float*)d_in[3];
    const float* ln1_b   = (const float*)d_in[4];
    const float* attn_w  = (const float*)d_in[5];
    const float* attn_b  = (const float*)d_in[6];
    const float* proj_w  = (const float*)d_in[7];
    const float* proj_b  = (const float*)d_in[8];
    const float* ln2_w   = (const float*)d_in[9];
    const float* ln2_b   = (const float*)d_in[10];
    const float* fc_w    = (const float*)d_in[11];
    const float* fc_b    = (const float*)d_in[12];
    const float* fcp_w   = (const float*)d_in[13];
    const float* fcp_b   = (const float*)d_in[14];
    const float* lnf_w   = (const float*)d_in[15];
    const float* lnf_b   = (const float*)d_in[16];
    const float* head_w  = (const float*)d_in[17];
    const float* head_b  = (const float*)d_in[18];

    const size_t NE = (size_t)BT * DD;     // 8M activations
    char* p = (char*)d_ws;
    float* x     = (float*)p;  p += NE * 4;            // residual stream (f32)
    bf16*  act   = (bf16*)p;   p += NE * 2;            // ln out / attn out (bf16)
    bf16*  qkvb  = (bf16*)p;   p += NE * 3 * 2;        // qkv bf16 (reused as lnf f32)
    bf16*  qh    = (bf16*)p;   p += NE * 2;
    bf16*  kh    = (bf16*)p;   p += NE * 2;
    bf16*  vth   = (bf16*)p;   p += NE * 2;
    bf16*  hbuf  = (bf16*)p;   p += NE * 2 * 2;        // [BT, 2D] bf16
    bf16*  wAttnT = (bf16*)p;  p += (size_t)LL * DD * 3 * DD * 2;  // [3D, D] per layer
    bf16*  wProjT = (bf16*)p;  p += (size_t)LL * DD * DD * 2;      // [D, D]
    bf16*  wFcT   = (bf16*)p;  p += (size_t)LL * DD * 2 * DD * 2;  // [2D, D]
    bf16*  wFcpT  = (bf16*)p;  p += (size_t)LL * 2 * DD * DD * 2;  // [D, 2D]
    float* lnfF  = (float*)qkvb;                       // reuse (33.5MB <= 50.3MB)

    // weights -> bf16, transposed to [N,K] for TDM-friendly B tiles
    for (int i = 0; i < LL; ++i) {
        long long nA = (long long)DD * 3 * DD;   // K=D,  N=3D
        long long nP = (long long)DD * DD;       // K=D,  N=D
        long long nF = (long long)DD * 2 * DD;   // K=D,  N=2D
        k_cvt_t<<<(unsigned)((nA + 255) / 256), 256, 0, stream>>>(
            attn_w + (size_t)i * nA, wAttnT + (size_t)i * nA, 3 * DD, DD);
        k_cvt_t<<<(unsigned)((nP + 255) / 256), 256, 0, stream>>>(
            proj_w + (size_t)i * nP, wProjT + (size_t)i * nP, DD, DD);
        k_cvt_t<<<(unsigned)((nF + 255) / 256), 256, 0, stream>>>(
            fc_w + (size_t)i * nF, wFcT + (size_t)i * nF, 2 * DD, DD);
        k_cvt_t<<<(unsigned)((nF + 255) / 256), 256, 0, stream>>>(
            fcp_w + (size_t)i * nF, wFcpT + (size_t)i * nF, DD, 2 * DD);
    }

    // x = X + wpe
    k_add_wpe<<<(unsigned)(NE / 256), 256, 0, stream>>>(X, wpe, x);

    const dim3 gQKV(3 * DD / 128, BT / 128);   // N=3072
    const dim3 gPRJ(DD / 128,     BT / 128);   // N=1024
    const dim3 gFC (2 * DD / 128, BT / 128);   // N=2048

    for (int i = 0; i < LL; ++i) {
        // x = relu(x); ln1 -> act (bf16)
        k_ln<true, false><<<BT, 256, 0, stream>>>(x, ln1_w + i * DD, ln1_b + i * DD,
                                                  act, nullptr);
        // qkv = relu(act @ attn_w + attn_b) -> bf16
        k_gemm<true, false, false, true><<<gQKV, 256, 0, stream>>>(
            act, wAttnT + (size_t)i * DD * 3 * DD, attn_b + (size_t)i * 3 * DD,
            nullptr, nullptr, qkvb, 3 * DD, DD);
        // split into per-head q, k, v^T
        k_split_qkv<<<(unsigned)(NE / 256), 256, 0, stream>>>(qkvb, qh, kh, vth);
        // causal relu attention -> act (bf16, [B,T,D])
        k_attn<<<(BB * HH * (TT / 32)) / 4, 128, 0, stream>>>(qh, kh, vth, act);
        // x += act @ proj_w + proj_b
        k_gemm<false, true, true, false><<<gPRJ, 256, 0, stream>>>(
            act, wProjT + (size_t)i * DD * DD, proj_b + (size_t)i * DD,
            x, x, nullptr, DD, DD);
        // ln2 -> act
        k_ln<false, false><<<BT, 256, 0, stream>>>(x, ln2_w + i * DD, ln2_b + i * DD,
                                                   act, nullptr);
        // h = relu(act @ fc_w + fc_b) -> bf16
        k_gemm<true, false, false, true><<<gFC, 256, 0, stream>>>(
            act, wFcT + (size_t)i * DD * 2 * DD, fc_b + (size_t)i * 2 * DD,
            nullptr, nullptr, hbuf, 2 * DD, DD);
        // x += h @ fc_proj_w + fc_proj_b
        k_gemm<false, true, true, false><<<gPRJ, 256, 0, stream>>>(
            hbuf, wFcpT + (size_t)i * 2 * DD * DD, fcp_b + (size_t)i * DD,
            x, x, nullptr, DD, 2 * DD);
    }

    // x = relu(x); lnf -> f32
    k_ln<true, true><<<BT, 256, 0, stream>>>(x, lnf_w, lnf_b, nullptr, lnfF);
    // emb = mean_t relu(lnf)  -> d_out[0 .. B*D)
    k_final<<<dim3(DD / 256, BB), 256, 0, stream>>>(lnfF, (float*)d_out);
    // head + loss -> d_out[B*D]
    k_head<<<1, 256, 0, stream>>>((const float*)d_out, head_w, head_b, Y,
                                  (float*)d_out + (size_t)BB * DD);
}
